// DotProductAttention_54795192762892
// MI455X (gfx1250) — matrix operand order repaired and checked by reading, not compiled, not run on hardware
//
#include <hip/hip_runtime.h>
#include <hip/hip_bf16.h>

// Flash-attention with per-element learnable scale, bf16 WMMA / fp32 accumulate.
// B=8, H=16, S=1024, D=128.  log2-domain softmax; row sums via ones-WMMA.

#define NB 8
#define NH 16
#define SEQ 1024
#define DH 128
#define BK 32                 // keys processed per LDS stage
#define WAVES 8               // waves per workgroup (wave32)
#define THREADS (WAVES * 32)
#define ROWS_BLK (WAVES * 16) // 128 query rows per workgroup

typedef __bf16 bf16_t;
typedef float  v8f   __attribute__((ext_vector_type(8)));
typedef bf16_t v16bf __attribute__((ext_vector_type(16)));
typedef bf16_t v8bf  __attribute__((ext_vector_type(8)));
typedef bf16_t v2bf  __attribute__((ext_vector_type(2)));

// LDS strides in elements; multiples of 16 so 16-element fragment reads are 32B aligned.
#define KSTRIDE (DH + 16)  // 144
#define VSTRIDE (BK + 16)  // 48
#define PSTRIDE (BK + 8)   // 40 (reads are 8-elem / 16B chunks)

// rsq(x * LN2SQ) == log2(e) * rsq(x): folds the exp->exp2 conversion into the rsqrt.
#define LN2SQ 0.4804530139182014f

__global__ __launch_bounds__(THREADS)
void fattn_scaled_kernel(const float* __restrict__ Q,
                         const float* __restrict__ K,
                         const float* __restrict__ V,
                         const float* __restrict__ Scale,
                         float* __restrict__ Out)
{
    __shared__ __align__(64) bf16_t Klds[BK * KSTRIDE];          // [key][d], row-major
    __shared__ __align__(64) bf16_t VTlds[DH * VSTRIDE];         // [d][key], transposed
    __shared__ __align__(64) bf16_t Plds[WAVES][16 * PSTRIDE];   // wave-private P staging

    const int tid  = threadIdx.x;
    const int wave = tid >> 5;
    const int lane = tid & 31;
    const int ln   = lane & 15;
    const int half = lane >> 4;

    const int h = blockIdx.y;
    const int b = blockIdx.z;
    const int q0 = blockIdx.x * ROWS_BLK + wave * 16;

    const size_t bh = (size_t)b * NH + h;
    const float* Qp    = Q + bh * SEQ * DH;
    const float* Kp    = K + bh * SEQ * DH;
    const float* Vp    = V + bh * SEQ * DH;
    const float* Sclp  = Scale + bh * (size_t)SEQ * SEQ;          // scale[b,h,:,:]  (streamed once)
    const float* Scl0p = Scale + (size_t)h * SEQ * SEQ;           // scale[0,h,:,:]  (L2-resident)
    float* Op = Out + bh * SEQ * DH;

    // Cooperative-load mapping: each thread stages two consecutive key rows x 8 d-columns.
    const int krow0 = (tid >> 4) * 2;      // even key row 0..30
    const int kcol0 = (tid & 15) * 8;      // d column 0..120

    // ---- Q fragments in A layout: lanes 0-15 = rows, K split across lane halves ----
    v16bf qa[4];
    {
        const float* qrow = Qp + (size_t)(q0 + ln) * DH;
        #pragma unroll
        for (int kf = 0; kf < 4; ++kf) {
            const int d0 = kf * 32 + 16 * half;
            #pragma unroll
            for (int i = 0; i < 8; ++i) {
                qa[kf][i]     = (bf16_t)qrow[d0 + i];       // K = kf*32 + 16*half + i
                qa[kf][i + 8] = (bf16_t)qrow[d0 + 8 + i];   // K = kf*32 + 16*half + 8 + i
            }
        }
    }

    // Constant ones B-fragment for row-sum WMMA.
    v16bf ones;
    #pragma unroll
    for (int i = 0; i < 16; ++i) ones[i] = (bf16_t)1.0f;

    v8f acc[8];
    v8f lacc = {};           // row sums of accumulated P (every column identical per row)
    float mrun[8];
    #pragma unroll
    for (int dt = 0; dt < 8; ++dt) acc[dt] = {};
    #pragma unroll
    for (int r = 0; r < 8; ++r) mrun[r] = -1e30f;   // log2-domain running max

    for (int kb = 0; kb < SEQ / BK; ++kb) {
        const int k0 = kb * BK;

        __syncthreads();  // previous stage fully consumed
        {
            const float* ksrc0 = Kp + (size_t)(k0 + krow0) * DH + kcol0;
            const float* ksrc1 = ksrc0 + DH;
            const float* vsrc0 = Vp + (size_t)(k0 + krow0) * DH + kcol0;
            const float* vsrc1 = vsrc0 + DH;
            // K tile: packed bf16 pairs along d (4-byte LDS stores)
            #pragma unroll
            for (int j = 0; j < 4; ++j) {
                v2bf a; a[0] = (bf16_t)ksrc0[2 * j]; a[1] = (bf16_t)ksrc0[2 * j + 1];
                v2bf c; c[0] = (bf16_t)ksrc1[2 * j]; c[1] = (bf16_t)ksrc1[2 * j + 1];
                *(v2bf*)&Klds[(krow0    ) * KSTRIDE + kcol0 + 2 * j] = a;
                *(v2bf*)&Klds[(krow0 + 1) * KSTRIDE + kcol0 + 2 * j] = c;
            }
            // V tile transposed: packed bf16 pairs along key (4-byte LDS stores)
            #pragma unroll
            for (int j = 0; j < 8; ++j) {
                v2bf pv; pv[0] = (bf16_t)vsrc0[j]; pv[1] = (bf16_t)vsrc1[j];
                *(v2bf*)&VTlds[(kcol0 + j) * VSTRIDE + krow0] = pv;
            }
        }
        __syncthreads();

        // ---- scores: two 16x16 C tiles covering 32 keys ----
        v8f st[2];
        #pragma unroll
        for (int nt = 0; nt < 2; ++nt) {
            v8f s = {};
            #pragma unroll
            for (int kf = 0; kf < 4; ++kf) {
                const v16bf kfrag =
                    *(const v16bf*)&Klds[(nt * 16 + ln) * KSTRIDE + kf * 32 + half * 16];
                s = __builtin_amdgcn_wmma_f32_16x16x32_bf16(
                        false, qa[kf], false, kfrag, (short)0, s, false, false);
            }
            // per-element learnable scaling, result in log2 units:
            //   s * scale * rsq(scale0 * ln2^2) == log2(e) * s * scale / sqrt(scale0)
            #pragma unroll
            for (int r = 0; r < 8; ++r) {
                const size_t off =
                    (size_t)(q0 + r + 8 * half) * SEQ + (size_t)(k0 + nt * 16 + ln);
                const float sc  = __builtin_nontemporal_load(&Sclp[off]); // single-use stream
                const float sc0 = Scl0p[off];                             // reused across batch
                s[r] = s[r] * sc * __builtin_amdgcn_rsqf(sc0 * LN2SQ);
            }
            st[nt] = s;
        }

        // ---- online softmax (log2 domain); max reduction within 16-lane halves ----
        float p0[8], p1[8];
        #pragma unroll
        for (int r = 0; r < 8; ++r) {
            float t = fmaxf(st[0][r], st[1][r]);
            t = fmaxf(t, __shfl_xor(t, 1, 32));
            t = fmaxf(t, __shfl_xor(t, 2, 32));
            t = fmaxf(t, __shfl_xor(t, 4, 32));
            t = fmaxf(t, __shfl_xor(t, 8, 32));
            const float mnew  = fmaxf(mrun[r], t);
            const float alpha = exp2f(mrun[r] - mnew);   // bare v_exp_f32
            p0[r] = exp2f(st[0][r] - mnew);
            p1[r] = exp2f(st[1][r] - mnew);
            mrun[r] = mnew;
            lacc[r] *= alpha;
            #pragma unroll
            for (int dt = 0; dt < 8; ++dt) acc[dt][r] *= alpha;
        }

        // ---- P: C layout -> A layout via wave-private LDS bounce ----
        #pragma unroll
        for (int r = 0; r < 8; ++r) {
            const int mrow = r + 8 * half;
            Plds[wave][mrow * PSTRIDE + ln]      = (bf16_t)p0[r];
            Plds[wave][mrow * PSTRIDE + 16 + ln] = (bf16_t)p1[r];
        }
        __builtin_amdgcn_wave_barrier();
        asm volatile("s_wait_dscnt 0" ::: "memory");  // same-wave cross-lane LDS visibility

        v16bf pfrag;
        {
            const v8bf lo = *(const v8bf*)&Plds[wave][ln * PSTRIDE + 16 * half];
            const v8bf hi = *(const v8bf*)&Plds[wave][ln * PSTRIDE + 16 * half + 8];
            pfrag = __builtin_shufflevector(lo, hi,
                        0, 1, 2, 3, 4, 5, 6, 7, 8, 9, 10, 11, 12, 13, 14, 15);
        }

        // ---- row sums: lacc += P x ones (replaces a 32-op shuffle/add tree) ----
        lacc = __builtin_amdgcn_wmma_f32_16x16x32_bf16(
                   false, pfrag, false, ones, (short)0, lacc, false, false);

        // ---- acc += P (16x32) x V (32x128), 8 WMMAs over D tiles ----
        #pragma unroll
        for (int dt = 0; dt < 8; ++dt) {
            const v16bf vfrag =
                *(const v16bf*)&VTlds[(dt * 16 + ln) * VSTRIDE + half * 16];
            acc[dt] = __builtin_amdgcn_wmma_f32_16x16x32_bf16(
                          false, pfrag, false, vfrag, (short)0, acc[dt], false, false);
        }
    }

    // ---- normalize and store (write-once -> non-temporal) ----
    #pragma unroll
    for (int r = 0; r < 8; ++r) {
        const float inv = 1.0f / lacc[r];   // all columns of lacc hold the row sum
        const size_t row = (size_t)(q0 + r + 8 * half) * DH;
        #pragma unroll
        for (int dt = 0; dt < 8; ++dt) {
            __builtin_nontemporal_store(acc[dt][r] * inv, &Op[row + dt * 16 + ln]);
        }
    }
}

extern "C" void kernel_launch(void* const* d_in, const int* in_sizes, int n_in,
                              void* d_out, int out_size, void* d_ws, size_t ws_size,
                              hipStream_t stream) {
    (void)in_sizes; (void)n_in; (void)out_size; (void)d_ws; (void)ws_size;
    const float* Q  = (const float*)d_in[0];
    const float* K  = (const float*)d_in[1];
    const float* V  = (const float*)d_in[2];
    const float* Sc = (const float*)d_in[3];
    float* Out = (float*)d_out;

    dim3 grid(SEQ / ROWS_BLK, NH, NB);  // (8, 16, 8)
    fattn_scaled_kernel<<<grid, THREADS, 0, stream>>>(Q, K, V, Sc, Out);
}
